// RouteNet_Fermi_59450937311992
// MI455X (gfx1250) — matrix operand order repaired and checked by reading, not compiled, not run on hardware
//
#include <hip/hip_runtime.h>
#include <hip/hip_bf16.h>

typedef __attribute__((ext_vector_type(16))) _Float16 v16h;
typedef __attribute__((ext_vector_type(8)))  _Float16 v8h;
typedef __attribute__((ext_vector_type(4)))  _Float16 v4h;
typedef __attribute__((ext_vector_type(8)))  float    v8f;

#define N_PATHS  100000
#define LSEQ     8
#define N_LINKS  10000
#define P_LINK   80
#define N_QUEUES 30000
#define P_Q      32
#define Q_LINK   3

// padded LDS row widths (halves) -> conflict-free b128 fragment loads
#define XROW 72   // 64 + 8  (144B stride: 36n mod 64 distinct for n=0..15)
#define HROW 40   // 32 + 8  (80B stride:  20n mod 64 distinct)
#define WKROW 72  // k-dim 64 (+8 pad)
#define WRROW 40  // k-dim 32 (+8 pad)

__device__ __forceinline__ float sigf(float x) { return 1.0f / (1.0f + __expf(-x)); }
__device__ __forceinline__ float zsf(float x, float m, float s) { return (x - m) / s; }

__device__ __forceinline__ v8f wmma_f16(v16h a, v16h b, v8f c) {
    return __builtin_amdgcn_wmma_f32_16x16x32_f16(false, a, false, b, (short)0, c, false, false);
}
// two 16B LDS loads -> one 16-half fragment (no scalar packing)
__device__ __forceinline__ v16h ld16(const _Float16* p0, const _Float16* p1) {
    v8h a = *(const v8h*)p0;
    v8h b = *(const v8h*)p1;
    return __builtin_shufflevector(a, b, 0,1,2,3,4,5,6,7,8,9,10,11,12,13,14,15);
}
__device__ __forceinline__ void st4h(_Float16* p, float4 v) {
    v4h h = { (_Float16)v.x, (_Float16)v.y, (_Float16)v.z, (_Float16)v.w };
    *(v4h*)p = h;
}

// ---------------------------------------------------------------------------
// Kernel A: path embedding MLP (17 -> 32 -> 32), init h_f/h_b/c_f/c_b
// ---------------------------------------------------------------------------
__global__ __launch_bounds__(256)
void path_emb_kernel(const float* __restrict__ traffic, const float* __restrict__ packets,
                     const float* __restrict__ eq_lambda, const float* __restrict__ avg_pkts_lambda,
                     const float* __restrict__ exp_max_factor, const float* __restrict__ pkts_lambda_on,
                     const float* __restrict__ avg_t_off, const float* __restrict__ avg_t_on,
                     const float* __restrict__ ar_a, const float* __restrict__ sigma,
                     const int* __restrict__ model,
                     const float* __restrict__ W1, const float* __restrict__ b1,
                     const float* __restrict__ W2, const float* __restrict__ b2,
                     float* __restrict__ hF, float* __restrict__ cF,
                     float* __restrict__ hB, float* __restrict__ cB)
{
    __shared__ float sW1[17 * 32], sW2[32 * 32], sb1[32], sb2[32];
    for (int i = threadIdx.x; i < 17 * 32; i += 256) sW1[i] = W1[i];
    for (int i = threadIdx.x; i < 32 * 32; i += 256) sW2[i] = W2[i];
    if (threadIdx.x < 32) { sb1[threadIdx.x] = b1[threadIdx.x]; sb2[threadIdx.x] = b2[threadIdx.x]; }
    __syncthreads();

    int p = blockIdx.x * 256 + threadIdx.x;
    if (p >= N_PATHS) return;

    float f[17];
    f[0] = zsf(traffic[p], 1385.4058837890625f, 859.8118896484375f);
    f[1] = zsf(packets[p], 1.4015231132507324f, 0.8932565450668335f);
    int md = model[p];
    #pragma unroll
    for (int k = 0; k < 7; ++k) f[2 + k] = (md == k) ? 1.0f : 0.0f;
    f[9]  = zsf(eq_lambda[p],       1350.97119140625f,    858.316162109375f);
    f[10] = zsf(avg_pkts_lambda[p], 0.9117304086685181f,  0.9723503589630127f);
    f[11] = zsf(exp_max_factor[p],  6.663637638092041f,   4.715115070343018f);
    f[12] = zsf(pkts_lambda_on[p],  0.9116322994232178f,  1.651275396347046f);
    f[13] = zsf(avg_t_off[p],       1.6649284362792969f,  2.356407403945923f);
    f[14] = zsf(avg_t_on[p],        1.6649284362792969f,  2.356407403945923f);
    f[15] = ar_a[p];
    f[16] = sigma[p];

    float h1[32];
    #pragma unroll
    for (int n = 0; n < 32; ++n) {
        float s = sb1[n];
        #pragma unroll
        for (int k = 0; k < 17; ++k) s += f[k] * sW1[k * 32 + n];
        h1[n] = fmaxf(s, 0.0f);
    }
    #pragma unroll
    for (int n = 0; n < 32; ++n) {
        float s = sb2[n];
        #pragma unroll
        for (int k = 0; k < 32; ++k) s += h1[k] * sW2[k * 32 + n];
        float o = fmaxf(s, 0.0f);
        if (n < 16) { hF[p * 16 + n] = o;        cF[p * 16 + n] = 0.0f; }
        else        { hB[p * 16 + n - 16] = o;   cB[p * 16 + n - 16] = 0.0f; }
    }
}

// ---------------------------------------------------------------------------
// Kernel B: link embedding (load gather-sum, MLP 5 -> 32 -> 32)
// ---------------------------------------------------------------------------
__global__ __launch_bounds__(256)
void link_emb_kernel(const float* __restrict__ traffic, const float* __restrict__ capacity,
                     const int* __restrict__ policy, const int* __restrict__ p2l,
                     const float* __restrict__ W1, const float* __restrict__ b1,
                     const float* __restrict__ W2, const float* __restrict__ b2,
                     float* __restrict__ linkH, float* __restrict__ linkC)
{
    __shared__ float sW1[5 * 32], sW2[32 * 32], sb1[32], sb2[32];
    for (int i = threadIdx.x; i < 5 * 32; i += 256) sW1[i] = W1[i];
    for (int i = threadIdx.x; i < 32 * 32; i += 256) sW2[i] = W2[i];
    if (threadIdx.x < 32) { sb1[threadIdx.x] = b1[threadIdx.x]; sb2[threadIdx.x] = b2[threadIdx.x]; }
    __syncthreads();

    int l = blockIdx.x * 256 + threadIdx.x;
    if (l >= N_LINKS) return;

    float load = 0.0f;
    for (int j = 0; j < P_LINK; ++j) load += traffic[p2l[(l * P_LINK + j) * 2]];
    load /= capacity[l];

    float f[5];
    f[0] = load;
    int pol = policy[l];
    #pragma unroll
    for (int k = 0; k < 4; ++k) f[1 + k] = (pol == k) ? 1.0f : 0.0f;

    float h1[32];
    #pragma unroll
    for (int n = 0; n < 32; ++n) {
        float s = sb1[n];
        #pragma unroll
        for (int k = 0; k < 5; ++k) s += f[k] * sW1[k * 32 + n];
        h1[n] = fmaxf(s, 0.0f);
    }
    #pragma unroll
    for (int n = 0; n < 32; ++n) {
        float s = sb2[n];
        #pragma unroll
        for (int k = 0; k < 32; ++k) s += h1[k] * sW2[k * 32 + n];
        linkH[l * 32 + n] = fmaxf(s, 0.0f);
        linkC[l * 32 + n] = 0.0f;
    }
}

// ---------------------------------------------------------------------------
// Kernel C: queue embedding (MLP 5 -> 32 -> 32)
// ---------------------------------------------------------------------------
__global__ __launch_bounds__(256)
void queue_emb_kernel(const float* __restrict__ queue_size, const int* __restrict__ priority,
                      const float* __restrict__ weight,
                      const float* __restrict__ W1, const float* __restrict__ b1,
                      const float* __restrict__ W2, const float* __restrict__ b2,
                      float* __restrict__ queueH, float* __restrict__ queueC)
{
    __shared__ float sW1[5 * 32], sW2[32 * 32], sb1[32], sb2[32];
    for (int i = threadIdx.x; i < 5 * 32; i += 256) sW1[i] = W1[i];
    for (int i = threadIdx.x; i < 32 * 32; i += 256) sW2[i] = W2[i];
    if (threadIdx.x < 32) { sb1[threadIdx.x] = b1[threadIdx.x]; sb2[threadIdx.x] = b2[threadIdx.x]; }
    __syncthreads();

    int q = blockIdx.x * 256 + threadIdx.x;
    if (q >= N_QUEUES) return;

    float f[5];
    f[0] = zsf(queue_size[q], 30259.10546875f, 21410.095703125f);
    int pr = priority[q];
    #pragma unroll
    for (int k = 0; k < 3; ++k) f[1 + k] = (pr == k) ? 1.0f : 0.0f;
    f[4] = weight[q];

    float h1[32];
    #pragma unroll
    for (int n = 0; n < 32; ++n) {
        float s = sb1[n];
        #pragma unroll
        for (int k = 0; k < 5; ++k) s += f[k] * sW1[k * 32 + n];
        h1[n] = fmaxf(s, 0.0f);
    }
    #pragma unroll
    for (int n = 0; n < 32; ++n) {
        float s = sb2[n];
        #pragma unroll
        for (int k = 0; k < 32; ++k) s += h1[k] * sW2[k * 32 + n];
        queueH[q * 32 + n] = fmaxf(s, 0.0f);
        queueC[q * 32 + n] = 0.0f;
    }
}

// ---------------------------------------------------------------------------
// Kernel D: bidirectional path LSTM over 8 steps. One wave = 16 paths.
// Weights transposed in LDS -> B fragments are 2x ds_load_b128 per tile.
// ---------------------------------------------------------------------------
__global__ __launch_bounds__(32)
void path_lstm_kernel(const float* __restrict__ gWkF, const float* __restrict__ gWrF, const float* __restrict__ gBF,
                      const float* __restrict__ gWkB, const float* __restrict__ gWrB, const float* __restrict__ gBB,
                      const int* __restrict__ q2p, const int* __restrict__ l2p,
                      const float* __restrict__ queueH, const float* __restrict__ linkH,
                      float* __restrict__ hF, float* __restrict__ cFst,
                      float* __restrict__ hB, float* __restrict__ cBst,
                      float* __restrict__ pss)
{
    __shared__ __align__(16) _Float16 sWkT[2][64][WKROW]; // [dir][n][k]
    __shared__ __align__(16) _Float16 sWrT[2][64][WRROW]; // [dir][n][k], k>=16 zero
    __shared__ float sB[2][64];
    __shared__ __align__(16) _Float16 sX[16][XROW];       // activations (16 paths x 64)
    __shared__ __align__(16) _Float16 sH[16][HROW];       // h tile, cols 16..31 zero

    const int lane  = threadIdx.x;
    const int pbase = blockIdx.x * 16;
    const int m16   = lane & 15;
    const int hi    = lane >> 4;

    for (int d = 0; d < 2; ++d) {
        const float* wk = d ? gWkB : gWkF;
        const float* wr = d ? gWrB : gWrF;
        const float* bb = d ? gBB  : gBF;
        for (int i = lane; i < 64 * 64; i += 32) {
            int n = i >> 6, k = i & 63;
            sWkT[d][n][k] = (_Float16)wk[k * 64 + n];
        }
        for (int i = lane; i < 64 * 32; i += 32) {
            int n = i >> 5, k = i & 31;
            sWrT[d][n][k] = (k < 16) ? (_Float16)wr[k * 64 + n] : (_Float16)0.0f;
        }
        for (int i = lane; i < 64; i += 32) sB[d][i] = bb[i];
    }

    // pss[:,0,:] = prev_h (state BEFORE this iteration's update)
    for (int i = lane; i < 16 * 32; i += 32) {
        int pi = i >> 5, dc = i & 31;
        float v = (dc < 16) ? hF[(size_t)(pbase + pi) * 16 + dc]
                            : hB[(size_t)(pbase + pi) * 16 + (dc - 16)];
        pss[((size_t)(pbase + pi) * 9 + 0) * 32 + dc] = v;
    }
    __builtin_amdgcn_wave_barrier();

    for (int dir = 0; dir < 2; ++dir) {
        const float* hSt = dir ? hB : hF;
        float* hStW      = dir ? hB : hF;
        const float* cSt = dir ? cBst : cFst;
        float* cStW      = dir ? cBst : cFst;

        for (int i = lane; i < 16 * 32; i += 32) {
            int pi = i >> 5, c = i & 31;
            sH[pi][c] = (c < 16) ? (_Float16)hSt[(size_t)(pbase + pi) * 16 + c] : (_Float16)0.0f;
        }
        float cReg[8];
        #pragma unroll
        for (int j = 0; j < 8; ++j)
            cReg[j] = cSt[(size_t)(pbase + j + hi * 8) * 16 + m16];
        __builtin_amdgcn_wave_barrier();

        for (int s = 0; s < 8; ++s) {
            const int ci  = dir ? (7 - s) : s;
            const int pos = dir ? (8 - s) : (s + 1);

            // gather x: lane covers (path m16, half hi): queue part or link part
            {
                int p   = pbase + m16;
                int idx = hi ? l2p[p * LSEQ + ci] : q2p[p * LSEQ + ci];
                const float4* src = (const float4*)((hi ? linkH : queueH) + (size_t)idx * 32);
                #pragma unroll
                for (int d = 0; d < 8; ++d)
                    st4h(&sX[m16][hi * 32 + d * 4], src[d]);
            }
            __builtin_amdgcn_wave_barrier();

            // A fragments: two b128 LDS loads each (documented 16-bit A layout)
            const int kb = hi * 8;
            v16h a0    = ld16(&sX[m16][kb],      &sX[m16][16 + kb]);
            v16h a1    = ld16(&sX[m16][32 + kb], &sX[m16][48 + kb]);
            v16h hfrag = ld16(&sH[m16][kb],      &sH[m16][16 + kb]);

            v8f acc[4];
            #pragma unroll
            for (int nt = 0; nt < 4; ++nt) {
                const int kb2 = hi * 8;
                const _Float16* wkRow = sWkT[dir][nt * 16 + m16];
                const _Float16* wrRow = sWrT[dir][nt * 16 + m16];
                v16h b0 = ld16(wkRow + kb2,      wkRow + kb2 + 16);
                v16h b1 = ld16(wkRow + 32 + kb2, wkRow + 32 + kb2 + 16);
                v16h br = ld16(wrRow + kb2,      wrRow + kb2 + 16);
                float bias = sB[dir][nt * 16 + m16];
                v8f c8;
                #pragma unroll
                for (int j = 0; j < 8; ++j) c8[j] = bias;
                c8 = wmma_f16(a0, b0, c8);
                c8 = wmma_f16(a1, b1, c8);
                c8 = wmma_f16(hfrag, br, c8);
                acc[nt] = c8;
            }
            __builtin_amdgcn_wave_barrier();

            // LSTM cell update in D-layout (elem j: path j+8*hi, gate-dim m16)
            #pragma unroll
            for (int j = 0; j < 8; ++j) {
                float iv = sigf(acc[0][j]);
                float fv = sigf(acc[1][j]);
                float gv = tanhf(acc[2][j]);
                float ov = sigf(acc[3][j]);
                float cc = fv * cReg[j] + iv * gv;
                float hv = ov * tanhf(cc);
                cReg[j] = cc;
                int mrow = j + hi * 8;
                sH[mrow][m16] = (_Float16)hv;
                pss[((size_t)(pbase + mrow) * 9 + pos) * 32 + dir * 16 + m16] = hv;
            }
            __builtin_amdgcn_wave_barrier();
        }

        // persist h, c for next message-passing iteration
        for (int i = lane; i < 16 * 16; i += 32) {
            int pi = i >> 4, d = i & 15;
            hStW[(size_t)(pbase + pi) * 16 + d] = (float)sH[pi][d];
        }
        #pragma unroll
        for (int j = 0; j < 8; ++j)
            cStW[(size_t)(pbase + j + hi * 8) * 16 + m16] = cReg[j];
        __builtin_amdgcn_wave_barrier();
    }
}

// ---------------------------------------------------------------------------
// Kernel E: queue update. path_sum gather (32 entries) + LSTM step 32->128.
// ---------------------------------------------------------------------------
__global__ __launch_bounds__(32)
void queue_update_kernel(const float* __restrict__ gWk, const float* __restrict__ gWr,
                         const float* __restrict__ gB,
                         const int* __restrict__ p2q, const float* __restrict__ pss,
                         float* __restrict__ queueH, float* __restrict__ queueC)
{
    __shared__ __align__(16) _Float16 sWkT[128][WRROW]; // [n][k], k<32
    __shared__ __align__(16) _Float16 sWrT[128][WRROW];
    __shared__ float sB[128];
    __shared__ __align__(16) _Float16 sX[16][HROW];
    __shared__ __align__(16) _Float16 sH[16][HROW];

    const int lane  = threadIdx.x;
    const int qbase = blockIdx.x * 16;
    const int m16   = lane & 15;
    const int hi    = lane >> 4;

    for (int i = lane; i < 128 * 32; i += 32) {
        int n = i >> 5, k = i & 31;
        sWkT[n][k] = (_Float16)gWk[k * 128 + n];
        sWrT[n][k] = (_Float16)gWr[k * 128 + n];
    }
    for (int i = lane; i < 128; i += 32) sB[i] = gB[i];

    // gather-sum path_sum: lane covers (queue m16, dims [hi*16, hi*16+16))
    {
        int q = qbase + m16;
        float acc[16];
        #pragma unroll
        for (int d = 0; d < 16; ++d) acc[d] = 0.0f;
        for (int j = 0; j < P_Q; ++j) {
            int p = p2q[(q * P_Q + j) * 2 + 0];
            int t = p2q[(q * P_Q + j) * 2 + 1];
            const float4* src = (const float4*)(pss + ((size_t)p * 9 + t) * 32 + hi * 16);
            #pragma unroll
            for (int d4 = 0; d4 < 4; ++d4) {
                float4 v = src[d4];
                acc[d4 * 4 + 0] += v.x; acc[d4 * 4 + 1] += v.y;
                acc[d4 * 4 + 2] += v.z; acc[d4 * 4 + 3] += v.w;
            }
        }
        #pragma unroll
        for (int d4 = 0; d4 < 4; ++d4) {
            float4 a4 = { acc[d4*4+0], acc[d4*4+1], acc[d4*4+2], acc[d4*4+3] };
            st4h(&sX[m16][hi * 16 + d4 * 4], a4);
        }
        const float4* hsrc = (const float4*)(queueH + (size_t)q * 32 + hi * 16);
        #pragma unroll
        for (int d4 = 0; d4 < 4; ++d4)
            st4h(&sH[m16][hi * 16 + d4 * 4], hsrc[d4]);
    }
    __builtin_amdgcn_wave_barrier();

    const int kb = hi * 8;
    v16h xf = ld16(&sX[m16][kb], &sX[m16][16 + kb]);
    v16h hf = ld16(&sH[m16][kb], &sH[m16][16 + kb]);

    v8f acc[8];
    #pragma unroll
    for (int nt = 0; nt < 8; ++nt) {
        const int kb2 = hi * 8;
        const _Float16* wkRow = sWkT[nt * 16 + m16];
        const _Float16* wrRow = sWrT[nt * 16 + m16];
        v16h bk = ld16(wkRow + kb2, wkRow + kb2 + 16);
        v16h br = ld16(wrRow + kb2, wrRow + kb2 + 16);
        float bias = sB[nt * 16 + m16];
        v8f c8;
        #pragma unroll
        for (int j = 0; j < 8; ++j) c8[j] = bias;
        c8 = wmma_f16(xf, bk, c8);
        c8 = wmma_f16(hf, br, c8);
        acc[nt] = c8;
    }

    // gates: i = nt0,1  f = nt2,3  g = nt4,5  o = nt6,7 (dim = half*16 + m16)
    #pragma unroll
    for (int half = 0; half < 2; ++half) {
        #pragma unroll
        for (int j = 0; j < 8; ++j) {
            size_t off = (size_t)(qbase + j + hi * 8) * 32 + half * 16 + m16;
            float cc = queueC[off];
            float iv = sigf(acc[0 + half][j]);
            float fv = sigf(acc[2 + half][j]);
            float gv = tanhf(acc[4 + half][j]);
            float ov = sigf(acc[6 + half][j]);
            cc = fv * cc + iv * gv;
            queueC[off] = cc;
            queueH[off] = ov * tanhf(cc);
        }
    }
}

// ---------------------------------------------------------------------------
// Kernel F: link update LSTM, 3 sequential steps over queue_to_link.
// ---------------------------------------------------------------------------
__global__ __launch_bounds__(32)
void link_update_kernel(const float* __restrict__ gWk, const float* __restrict__ gWr,
                        const float* __restrict__ gB,
                        const int* __restrict__ q2l, const float* __restrict__ queueH,
                        float* __restrict__ linkH, float* __restrict__ linkC)
{
    __shared__ __align__(16) _Float16 sWkT[128][WRROW];
    __shared__ __align__(16) _Float16 sWrT[128][WRROW];
    __shared__ float sB[128];
    __shared__ __align__(16) _Float16 sX[16][HROW];
    __shared__ __align__(16) _Float16 sH[16][HROW];

    const int lane  = threadIdx.x;
    const int lbase = blockIdx.x * 16;
    const int m16   = lane & 15;
    const int hi    = lane >> 4;

    for (int i = lane; i < 128 * 32; i += 32) {
        int n = i >> 5, k = i & 31;
        sWkT[n][k] = (_Float16)gWk[k * 128 + n];
        sWrT[n][k] = (_Float16)gWr[k * 128 + n];
    }
    for (int i = lane; i < 128; i += 32) sB[i] = gB[i];

    {
        const float4* hsrc = (const float4*)(linkH + (size_t)(lbase + m16) * 32 + hi * 16);
        #pragma unroll
        for (int d4 = 0; d4 < 4; ++d4)
            st4h(&sH[m16][hi * 16 + d4 * 4], hsrc[d4]);
    }
    float cReg[2][8];
    #pragma unroll
    for (int half = 0; half < 2; ++half)
        #pragma unroll
        for (int j = 0; j < 8; ++j)
            cReg[half][j] = linkC[(size_t)(lbase + j + hi * 8) * 32 + half * 16 + m16];
    __builtin_amdgcn_wave_barrier();

    for (int s = 0; s < Q_LINK; ++s) {
        {
            int q = q2l[(lbase + m16) * Q_LINK + s];
            const float4* src = (const float4*)(queueH + (size_t)q * 32 + hi * 16);
            #pragma unroll
            for (int d4 = 0; d4 < 4; ++d4)
                st4h(&sX[m16][hi * 16 + d4 * 4], src[d4]);
        }
        __builtin_amdgcn_wave_barrier();

        const int kb = hi * 8;
        v16h xf = ld16(&sX[m16][kb], &sX[m16][16 + kb]);
        v16h hf = ld16(&sH[m16][kb], &sH[m16][16 + kb]);

        v8f acc[8];
        #pragma unroll
        for (int nt = 0; nt < 8; ++nt) {
            const int kb2 = hi * 8;
            const _Float16* wkRow = sWkT[nt * 16 + m16];
            const _Float16* wrRow = sWrT[nt * 16 + m16];
            v16h bk = ld16(wkRow + kb2, wkRow + kb2 + 16);
            v16h br = ld16(wrRow + kb2, wrRow + kb2 + 16);
            float bias = sB[nt * 16 + m16];
            v8f c8;
            #pragma unroll
            for (int j = 0; j < 8; ++j) c8[j] = bias;
            c8 = wmma_f16(xf, bk, c8);
            c8 = wmma_f16(hf, br, c8);
            acc[nt] = c8;
        }
        __builtin_amdgcn_wave_barrier();

        #pragma unroll
        for (int half = 0; half < 2; ++half) {
            #pragma unroll
            for (int j = 0; j < 8; ++j) {
                float cc = cReg[half][j];
                float iv = sigf(acc[0 + half][j]);
                float fv = sigf(acc[2 + half][j]);
                float gv = tanhf(acc[4 + half][j]);
                float ov = sigf(acc[6 + half][j]);
                cc = fv * cc + iv * gv;
                cReg[half][j] = cc;
                sH[j + hi * 8][half * 16 + m16] = (_Float16)(ov * tanhf(cc));
            }
        }
        __builtin_amdgcn_wave_barrier();
    }

    for (int i = lane; i < 16 * 32; i += 32)
        linkH[(size_t)(lbase + (i >> 5)) * 32 + (i & 31)] = (float)sH[i >> 5][i & 31];
    #pragma unroll
    for (int half = 0; half < 2; ++half)
        #pragma unroll
        for (int j = 0; j < 8; ++j)
            linkC[(size_t)(lbase + j + hi * 8) * 32 + half * 16 + m16] = cReg[half][j];
}

// ---------------------------------------------------------------------------
// Kernel G: readout MLP (32->16->16->1) over pss[:,1:], delay accumulation
// ---------------------------------------------------------------------------
__global__ __launch_bounds__(256)
void readout_kernel(const float* __restrict__ pss, const float* __restrict__ capacity,
                    const int* __restrict__ l2p,
                    const float* __restrict__ traffic, const float* __restrict__ packets,
                    const float* __restrict__ W1, const float* __restrict__ b1,
                    const float* __restrict__ W2, const float* __restrict__ b2,
                    const float* __restrict__ W3, const float* __restrict__ b3,
                    float* __restrict__ out)
{
    __shared__ float sW1[32 * 16], sW2[16 * 16], sW3[16], sb1[16], sb2[16], sb3;
    for (int i = threadIdx.x; i < 32 * 16; i += 256) sW1[i] = W1[i];
    for (int i = threadIdx.x; i < 16 * 16; i += 256) sW2[i] = W2[i];
    if (threadIdx.x < 16) { sW3[threadIdx.x] = W3[threadIdx.x];
                            sb1[threadIdx.x] = b1[threadIdx.x];
                            sb2[threadIdx.x] = b2[threadIdx.x]; }
    if (threadIdx.x == 0) sb3 = b3[0];
    __syncthreads();

    int p = blockIdx.x * 256 + threadIdx.x;
    if (p >= N_PATHS) return;

    float qd = 0.0f, sinv = 0.0f;
    for (int t = 0; t < LSEQ; ++t) {
        const float* xr = pss + ((size_t)p * 9 + t + 1) * 32;
        float h1[16];
        #pragma unroll
        for (int n = 0; n < 16; ++n) {
            float s = sb1[n];
            #pragma unroll
            for (int k = 0; k < 32; ++k) s += xr[k] * sW1[k * 16 + n];
            h1[n] = fmaxf(s, 0.0f);
        }
        float occ = sb3;
        #pragma unroll
        for (int n = 0; n < 16; ++n) {
            float s = sb2[n];
            #pragma unroll
            for (int k = 0; k < 16; ++k) s += h1[k] * sW2[k * 16 + n];
            occ += fmaxf(s, 0.0f) * sW3[n];
        }
        float cap = capacity[l2p[p * LSEQ + t]];
        qd   += occ / cap;
        sinv += 1.0f / cap;
    }
    out[p] = qd + (traffic[p] / packets[p]) * sinv;
}

// ---------------------------------------------------------------------------
extern "C" void kernel_launch(void* const* d_in, const int* in_sizes, int n_in,
                              void* d_out, int out_size, void* d_ws, size_t ws_size,
                              hipStream_t stream) {
    auto F = [&](int i) { return (const float*)d_in[i]; };
    auto I = [&](int i) { return (const int*)d_in[i]; };

    const float* traffic        = F(0);
    const float* packets        = F(1);
    const float* eq_lambda      = F(2);
    const float* avg_pkts_la    = F(3);
    const float* exp_max_factor = F(4);
    const float* pkts_lambda_on = F(5);
    const float* avg_t_off      = F(6);
    const float* avg_t_on       = F(7);
    const float* ar_a           = F(8);
    const float* sigma          = F(9);
    const float* capacity       = F(10);
    const float* queue_size     = F(11);
    const float* weight         = F(12);
    const int*   model          = I(13);
    const int*   policy         = I(14);
    const int*   priority       = I(15);
    /* 16: length (constant L, unused) */
    const int*   q2p            = I(17);
    const int*   l2p            = I(18);
    const int*   p2l            = I(19);
    const int*   p2q            = I(20);
    const int*   q2l            = I(21);

    // params flattened in JAX pytree (sorted dict key) order
    const float* le_W1 = F(22); const float* le_W2 = F(23);
    const float* le_b1 = F(24); const float* le_b2 = F(25);
    const float* lu_Wk = F(26); const float* lu_Wr = F(27); const float* lu_b = F(28);
    const float* pb_Wk = F(29); const float* pb_Wr = F(30); const float* pb_b = F(31);
    const float* pe_W1 = F(32); const float* pe_W2 = F(33);
    const float* pe_b1 = F(34); const float* pe_b2 = F(35);
    const float* pf_Wk = F(36); const float* pf_Wr = F(37); const float* pf_b = F(38);
    const float* qe_W1 = F(39); const float* qe_W2 = F(40);
    const float* qe_b1 = F(41); const float* qe_b2 = F(42);
    const float* qu_Wk = F(43); const float* qu_Wr = F(44); const float* qu_b = F(45);
    const float* ro_W1 = F(46); const float* ro_W2 = F(47); const float* ro_W3 = F(48);
    const float* ro_b1 = F(49); const float* ro_b2 = F(50); const float* ro_b3 = F(51);

    // workspace carve-up (all offsets 256B aligned)
    char* ws = (char*)d_ws;
    float* pss = (float*)(ws);                    // 100000*9*32*4 = 115,200,000
    float* hF  = (float*)(ws + 115200000);        // 100000*16*4  =   6,400,000
    float* cF  = (float*)(ws + 121600000);
    float* hB  = (float*)(ws + 128000000);
    float* cB  = (float*)(ws + 134400000);
    float* qH  = (float*)(ws + 140800000);        // 30000*32*4   =   3,840,000
    float* qC  = (float*)(ws + 144640000);
    float* lH  = (float*)(ws + 148480000);        // 10000*32*4   =   1,280,000
    float* lC  = (float*)(ws + 149760000);        // end 151,040,000

    path_emb_kernel<<<(N_PATHS + 255) / 256, 256, 0, stream>>>(
        traffic, packets, eq_lambda, avg_pkts_la, exp_max_factor, pkts_lambda_on,
        avg_t_off, avg_t_on, ar_a, sigma, model,
        pe_W1, pe_b1, pe_W2, pe_b2, hF, cF, hB, cB);
    link_emb_kernel<<<(N_LINKS + 255) / 256, 256, 0, stream>>>(
        traffic, capacity, policy, p2l, le_W1, le_b1, le_W2, le_b2, lH, lC);
    queue_emb_kernel<<<(N_QUEUES + 255) / 256, 256, 0, stream>>>(
        queue_size, priority, weight, qe_W1, qe_b1, qe_W2, qe_b2, qH, qC);

    for (int it = 0; it < 8; ++it) {
        path_lstm_kernel<<<N_PATHS / 16, 32, 0, stream>>>(
            pf_Wk, pf_Wr, pf_b, pb_Wk, pb_Wr, pb_b,
            q2p, l2p, qH, lH, hF, cF, hB, cB, pss);
        queue_update_kernel<<<N_QUEUES / 16, 32, 0, stream>>>(
            qu_Wk, qu_Wr, qu_b, p2q, pss, qH, qC);
        link_update_kernel<<<N_LINKS / 16, 32, 0, stream>>>(
            lu_Wk, lu_Wr, lu_b, q2l, qH, lH, lC);
    }

    readout_kernel<<<(N_PATHS + 255) / 256, 256, 0, stream>>>(
        pss, capacity, l2p, traffic, packets,
        ro_W1, ro_b1, ro_W2, ro_b2, ro_W3, ro_b3, (float*)d_out);
}
